// OptimizedVQGAN_14929306320977
// MI455X (gfx1250) — matrix operand order repaired and checked by reading, not compiled, not run on hardware
//
#include <hip/hip_runtime.h>
#include <hip/hip_bf16.h>
#include <math.h>

// ---------------------------------------------------------------------------
// CDNA5 (gfx1250) VQGAN forward.
//  * Convs (implicit GEMM) + attention GEMMs on v_wmma_f32_16x16x32_f16,
//    register-blocked: 4 WMMA per wave per K-step (32x32 C tile per wave).
//  * Attention AV-GEMM stages its B tile with the Tensor Data Mover
//    (tensor_load_to_lds + s_wait_tensorcnt), 6-arg clang-23 builtin form.
//  * Elementwise stages (GN/SiLU/softmax/VQ/upsample) stay f32, BW-bound.
// ---------------------------------------------------------------------------

typedef __attribute__((ext_vector_type(16))) _Float16 v16h;
typedef __attribute__((ext_vector_type(8)))  float    v8f;
typedef __attribute__((ext_vector_type(4)))  unsigned int u32x4;
typedef __attribute__((ext_vector_type(8)))  int i32x8;
typedef __attribute__((ext_vector_type(4)))  int i32x4;

#define TM2 64     // macro-tile rows (2 waves x 32)
#define TN2 128    // macro-tile cols (4 waves x 32)
#define KS  32     // K step (one WMMA K depth)
#define LDA_H 36   // LDS row stride (halves) for A tile
#define LDB_H 36   // LDS row stride (halves) for f16 B tile

// ---- fragment loaders (CDNA5 16-bit A 16x32 / B 32x16 lane layouts) --------
__device__ __forceinline__ v16h frag_a(const _Float16* As, int row, int laneHalf) {
  v16h a;
#pragma unroll
  for (int i = 0; i < 16; ++i) {
    int kl = (laneHalf << 4) + i;   // K within 0..31
    a[i] = As[row * LDA_H + kl];
  }
  return a;
}
__device__ __forceinline__ v16h frag_b16(const _Float16* Bs, int col, int laneHalf) {
  v16h b;
#pragma unroll
  for (int i = 0; i < 16; ++i)
    b[i] = Bs[col * LDB_H + (laneHalf << 4) + i];            // K = laneHalf*16 + i
  return b;
}
__device__ __forceinline__ v16h frag_b32(const float* Bsf, int col, int laneHalf) {
  v16h b;
#pragma unroll
  for (int i = 0; i < 16; ++i)
    b[i] = (_Float16)Bsf[col * KS + (laneHalf << 4) + i];
  return b;
}

// ---------------------------------------------------------------------------
// Implicit-GEMM convolution: M = Cout, N = Ho*Wo (per image), K = Cin*k*k.
// ---------------------------------------------------------------------------
__global__ void __launch_bounds__(256) vqgan_conv_wmma(
    const float* __restrict__ in, const float* __restrict__ wgt,
    float* __restrict__ out,
    int Cin, int H, int W, int Cout, int Ho, int Wo,
    int KW, int stride, int pad)
{
  __shared__ _Float16 As[TM2 * LDA_H];
  __shared__ _Float16 Bs[TN2 * LDB_H];

  const int tid  = threadIdx.x;
  const int lane = tid & 31;
  const int wave = tid >> 5;
  const int wm = wave >> 2;          // 0..1  (M sub-tile)
  const int wn = wave & 3;           // 0..3  (N sub-tile)
  const int laneHalf = lane >> 4;
  const int lr = lane & 15;

  const int HoWo = Ho * Wo;
  const int K = Cin * KW * KW;
  const int nBase = blockIdx.x * TN2;
  const int mBase = blockIdx.y * TM2;
  const int nb = blockIdx.z;
  const size_t inBase = (size_t)nb * Cin * H * W;

  v8f a00 = {}, a01 = {}, a10 = {}, a11 = {};

  for (int k0 = 0; k0 < K; k0 += KS) {
    // ---- stage A (weights) tile, f32 -> f16 ----
    for (int i = tid; i < TM2 * KS; i += 256) {
      int m = i >> 5, kk = i & (KS - 1);
      int gm = mBase + m, gk = k0 + kk;
      float v = (gm < Cout && gk < K) ? wgt[(size_t)gm * K + gk] : 0.f;
      if (gm < Cout && gk + KS < K)
        __builtin_prefetch(wgt + (size_t)gm * K + gk + KS, 0, 1);
      As[m * LDA_H + kk] = (_Float16)v;
    }
    // ---- stage B (im2col of input) tile ----
    for (int i = tid; i < TN2 * KS; i += 256) {
      int col = i >> 5, kk = i & (KS - 1);
      int gc = nBase + col, gk = k0 + kk;
      float v = 0.f;
      if (gc < HoWo && gk < K) {
        int cin, ry, rx;
        if (KW == 3) { cin = gk / 9; int r = gk - cin * 9; ry = r / 3; rx = r - ry * 3; }
        else         { cin = gk; ry = 0; rx = 0; }
        int oy = gc / Wo, ox = gc - oy * Wo;
        int iy = oy * stride - pad + ry;
        int ix = ox * stride - pad + rx;
        if ((unsigned)iy < (unsigned)H && (unsigned)ix < (unsigned)W)
          v = in[inBase + ((size_t)cin * H + iy) * W + ix];
      }
      Bs[col * LDB_H + kk] = (_Float16)v;
    }
    __syncthreads();

    v16h af0 = frag_a(As, wm * 32 + lr,      laneHalf);
    v16h af1 = frag_a(As, wm * 32 + 16 + lr, laneHalf);
    v16h bf0 = frag_b16(Bs, wn * 32 + lr,      laneHalf);
    v16h bf1 = frag_b16(Bs, wn * 32 + 16 + lr, laneHalf);
    a00 = __builtin_amdgcn_wmma_f32_16x16x32_f16(false, af0, false, bf0, (short)0, a00, false, false);
    a01 = __builtin_amdgcn_wmma_f32_16x16x32_f16(false, af0, false, bf1, (short)0, a01, false, false);
    a10 = __builtin_amdgcn_wmma_f32_16x16x32_f16(false, af1, false, bf0, (short)0, a10, false, false);
    a11 = __builtin_amdgcn_wmma_f32_16x16x32_f16(false, af1, false, bf1, (short)0, a11, false, false);
    __syncthreads();
  }

  // ---- store C (VGPR r -> M = r + 8*laneHalf; N = lane&15) ----
  const size_t outBase = (size_t)nb * Cout * HoWo;
#pragma unroll
  for (int ni = 0; ni < 2; ++ni) {
    int n = nBase + wn * 32 + ni * 16 + lr;
    if (n >= HoWo) continue;
#pragma unroll
    for (int mi = 0; mi < 2; ++mi) {
      v8f acc = (mi == 0) ? (ni == 0 ? a00 : a01) : (ni == 0 ? a10 : a11);
#pragma unroll
      for (int r = 0; r < 8; ++r) {
        int m = mBase + wm * 32 + mi * 16 + r + laneHalf * 8;
        if (m < Cout)
          out[outBase + (size_t)m * HoWo + n] = acc[r];
      }
    }
  }
}

// ---------------------------------------------------------------------------
// Batched GEMM: C = scale * op(A)[MxK] * op(B)[KxN], f32 in/out, f16 WMMA.
// opA: A(m,k)=A[k*lda+m] if 1.   opB: B(k,n)=B[n*ldb+k] if 1.
// useTdm (requires opB==1, N%TN2==0, K%KS==0): B tile staged by the Tensor
// Data Mover into LDS as f32 (2D tile: TN2 rows of KS contiguous elements,
// row stride ldb), converted f32->f16 during fragment load.
// ---------------------------------------------------------------------------
__global__ void __launch_bounds__(256) vqgan_gemm_wmma(
    const float* __restrict__ A, const float* __restrict__ B,
    float* __restrict__ C,
    int M, int N, int K, int lda, int ldb, int ldc,
    long sA, long sB, long sC, int opA, int opB, float scale, int useTdm)
{
  __shared__ _Float16 As[TM2 * LDA_H];
  __shared__ float    Bsf[TN2 * KS];

  A += (size_t)blockIdx.z * sA;
  B += (size_t)blockIdx.z * sB;
  C += (size_t)blockIdx.z * sC;

  const int tid  = threadIdx.x;
  const int lane = tid & 31;
  const int wave = tid >> 5;
  const int wm = wave >> 2, wn = wave & 3;
  const int laneHalf = lane >> 4;
  const int lr = lane & 15;
  const int nBase = blockIdx.x * TN2;
  const int mBase = blockIdx.y * TM2;

  v8f a00 = {}, a01 = {}, a10 = {}, a11 = {};

  for (int k0 = 0; k0 < K; k0 += KS) {
    // ---- stage A tile, f32 -> f16 ----
    for (int i = tid; i < TM2 * KS; i += 256) {
      int m = i >> 5, kk = i & (KS - 1);
      int gm = mBase + m, gk = k0 + kk;
      float v = 0.f;
      if (gm < M && gk < K)
        v = opA ? A[(size_t)gk * lda + gm] : A[(size_t)gm * lda + gk];
      As[m * LDA_H + kk] = (_Float16)v;
    }
    // ---- stage B tile (TDM or cooperative) ----
    if (useTdm) {
      if (wave == 0) {
        // D# group 0: count=1, lds_addr, global_addr(57b), type=2
        unsigned ldsAddr = (unsigned)(unsigned long long)(size_t)&Bsf[0];
        unsigned long long ga =
            (unsigned long long)(size_t)(B + (size_t)nBase * ldb + k0);
        u32x4 g0;
        g0[0] = 1u;
        g0[1] = ldsAddr;
        g0[2] = (unsigned)(ga & 0xffffffffull);
        g0[3] = (unsigned)((ga >> 32) & 0x01ffffffull) | (2u << 30);
        // D# group 1: data_size=4B; tensor dims huge (no OOB clip);
        // tile_dim0=KS, tile_dim1=TN2; tensor_dim0_stride=ldb.
        unsigned t0 = 0x7fffffffu, t1 = 0x7fffffffu;
        unsigned long long st0 = (unsigned long long)(unsigned)ldb;
        i32x8 g1;
        g1[0] = (int)(2u << 16);
        g1[1] = (int)((t0 & 0xffffu) << 16);
        g1[2] = (int)(((t0 >> 16) & 0xffffu) | ((t1 & 0xffffu) << 16));
        g1[3] = (int)(((t1 >> 16) & 0xffffu) | ((unsigned)KS << 16));
        g1[4] = (int)(unsigned)TN2;                 // tile_dim1=TN2, tile_dim2=0
        g1[5] = (int)(unsigned)(st0 & 0xffffffffull);
        g1[6] = (int)(unsigned)((st0 >> 32) & 0xffffull);
        g1[7] = 0;
        i32x4 gz4 = {0, 0, 0, 0};
        i32x8 gz8 = {0, 0, 0, 0, 0, 0, 0, 0};
        __builtin_amdgcn_tensor_load_to_lds(g0, g1, gz4, gz4, gz8, 0);
        __builtin_amdgcn_s_wait_tensorcnt(0);
      }
    } else {
      for (int i = tid; i < TN2 * KS; i += 256) {
        int col = i >> 5, kk = i & (KS - 1);
        int gc = nBase + col, gk = k0 + kk;
        float v = 0.f;
        if (gc < N && gk < K)
          v = opB ? B[(size_t)gc * ldb + gk] : B[(size_t)gk * ldb + gc];
        Bsf[col * KS + kk] = v;
      }
    }
    __syncthreads();

    v16h af0 = frag_a(As, wm * 32 + lr,      laneHalf);
    v16h af1 = frag_a(As, wm * 32 + 16 + lr, laneHalf);
    v16h bf0 = frag_b32(Bsf, wn * 32 + lr,      laneHalf);
    v16h bf1 = frag_b32(Bsf, wn * 32 + 16 + lr, laneHalf);
    a00 = __builtin_amdgcn_wmma_f32_16x16x32_f16(false, af0, false, bf0, (short)0, a00, false, false);
    a01 = __builtin_amdgcn_wmma_f32_16x16x32_f16(false, af0, false, bf1, (short)0, a01, false, false);
    a10 = __builtin_amdgcn_wmma_f32_16x16x32_f16(false, af1, false, bf0, (short)0, a10, false, false);
    a11 = __builtin_amdgcn_wmma_f32_16x16x32_f16(false, af1, false, bf1, (short)0, a11, false, false);
    __syncthreads();
  }

#pragma unroll
  for (int ni = 0; ni < 2; ++ni) {
    int n = nBase + wn * 32 + ni * 16 + lr;
    if (n >= N) continue;
#pragma unroll
    for (int mi = 0; mi < 2; ++mi) {
      v8f acc = (mi == 0) ? (ni == 0 ? a00 : a01) : (ni == 0 ? a10 : a11);
#pragma unroll
      for (int r = 0; r < 8; ++r) {
        int m = mBase + wm * 32 + mi * 16 + r + laneHalf * 8;
        if (m < M)
          C[(size_t)m * ldc + n] = scale * acc[r];
      }
    }
  }
}

// ---------------------------------------------------------------------------
// GroupNorm (two pass), fused apply with optional residual add and SiLU.
// ---------------------------------------------------------------------------
__global__ void __launch_bounds__(256) vqgan_gn_stats(
    const float* __restrict__ x, float* __restrict__ stats,
    int C, int HW, int groups)
{
  __shared__ float rs[256], rss[256];
  int ng = blockIdx.x;
  int g = ng % groups, n = ng / groups;
  int cpg = C / groups;
  size_t base = ((size_t)n * C + (size_t)g * cpg) * HW;
  long M = (long)cpg * HW;
  float s = 0.f, ss = 0.f;
  for (long i = threadIdx.x; i < M; i += 256) {
    float v = x[base + i];
    s += v; ss += v * v;
  }
  rs[threadIdx.x] = s; rss[threadIdx.x] = ss;
  __syncthreads();
  for (int st = 128; st > 0; st >>= 1) {
    if ((int)threadIdx.x < st) {
      rs[threadIdx.x]  += rs[threadIdx.x + st];
      rss[threadIdx.x] += rss[threadIdx.x + st];
    }
    __syncthreads();
  }
  if (threadIdx.x == 0) {
    float mu  = rs[0] / (float)M;
    float var = rss[0] / (float)M - mu * mu;
    stats[2 * ng]     = mu;
    stats[2 * ng + 1] = rsqrtf(var + 1e-5f);
  }
}

// mode bit0: SiLU, bit1: += add[]
__global__ void vqgan_gn_apply(
    const float* __restrict__ x, const float* __restrict__ stats,
    const float* __restrict__ gm, const float* __restrict__ bt,
    const float* __restrict__ add, float* __restrict__ y,
    int C, int HW, int groups, int mode, long total)
{
  long i = (long)blockIdx.x * blockDim.x + threadIdx.x;
  if (i >= total) return;
  int c = (int)((i / HW) % C);
  int n = (int)(i / ((long)HW * C));
  int g = c / (C / groups);
  int sg = 2 * (n * groups + g);
  float v = (x[i] - stats[sg]) * stats[sg + 1] * gm[c] + bt[c];
  if (mode & 2) v += add[i];
  if (mode & 1) v *= 1.f / (1.f + __expf(-v));
  y[i] = v;
}

__global__ void vqgan_add(const float* __restrict__ a, const float* __restrict__ b,
                          float* __restrict__ y, long total)
{
  long i = (long)blockIdx.x * blockDim.x + threadIdx.x;
  if (i < total) y[i] = a[i] + b[i];
}

__global__ void vqgan_up2(const float* __restrict__ in, float* __restrict__ out,
                          int H, int W, long total)   // total = out elems
{
  long i = (long)blockIdx.x * blockDim.x + threadIdx.x;
  if (i >= total) return;
  int W2 = W * 2;
  long hw2 = (long)H * 2 * W2;
  long pc = i / hw2;                       // n*C + c
  int rem = (int)(i - pc * hw2);
  int y2 = rem / W2, x2 = rem - y2 * W2;
  out[i] = in[pc * (long)H * W + (long)(y2 >> 1) * W + (x2 >> 1)];
}

__global__ void vqgan_bias_sigmoid(const float* __restrict__ x,
                                   const float* __restrict__ bias,
                                   float* __restrict__ y,
                                   int C, int HW, long total)
{
  long i = (long)blockIdx.x * blockDim.x + threadIdx.x;
  if (i >= total) return;
  int c = (int)((i / HW) % C);
  float v = x[i] + bias[c];
  y[i] = 1.f / (1.f + __expf(-v));
}

__global__ void vqgan_softmax(float* __restrict__ x, int L)
{
  __shared__ float red[256];
  size_t base = (size_t)blockIdx.x * L;
  int tid = threadIdx.x;
  float m = -3.4e38f;
  for (int i = tid; i < L; i += 256) m = fmaxf(m, x[base + i]);
  red[tid] = m; __syncthreads();
  for (int s = 128; s > 0; s >>= 1) {
    if (tid < s) red[tid] = fmaxf(red[tid], red[tid + s]);
    __syncthreads();
  }
  float mx = red[0]; __syncthreads();
  float sum = 0.f;
  for (int i = tid; i < L; i += 256) {
    float e = __expf(x[base + i] - mx);
    x[base + i] = e;
    sum += e;
  }
  red[tid] = sum; __syncthreads();
  for (int s = 128; s > 0; s >>= 1) {
    if (tid < s) red[tid] += red[tid + s];
    __syncthreads();
  }
  float inv = 1.f / red[0];
  for (int i = tid; i < L; i += 256) x[base + i] *= inv;
}

// VQ: 4096 vectors of dim 16 (taken along W, per torch view(-1, D));
// forward value of straight-through is simply the nearest codebook vector.
__global__ void vqgan_vq(const float* __restrict__ z, const float* __restrict__ emb,
                         float* __restrict__ q, int rows, int E, int D)
{
  int r = blockIdx.x * blockDim.x + threadIdx.x;
  if (r >= rows) return;
  float zv[16];
  for (int j = 0; j < D; ++j) zv[j] = z[(size_t)r * D + j];
  int best = 0; float bestd = 3.4e38f;
  for (int e = 0; e < E; ++e) {
    float dot = 0.f, nn = 0.f;
    for (int j = 0; j < D; ++j) {
      float ev = emb[(size_t)e * D + j];
      dot += zv[j] * ev; nn += ev * ev;
    }
    float d = nn - 2.f * dot;          // |z|^2 constant per row -> same argmin
    if (d < bestd) { bestd = d; best = e; }
  }
  for (int j = 0; j < D; ++j) q[(size_t)r * D + j] = emb[(size_t)best * D + j];
}

// ---------------------------------------------------------------------------
// Host orchestration
// ---------------------------------------------------------------------------
extern "C" void kernel_launch(void* const* d_in, const int* in_sizes, int n_in,
                              void* d_out, int out_size, void* d_ws, size_t ws_size,
                              hipStream_t stream)
{
  (void)in_sizes; (void)n_in; (void)out_size; (void)ws_size;
  auto I = [&](int i) { return (const float*)d_in[i]; };

  const int NB = 16;
  float* ws = (float*)d_ws;
  const long PSL = 16L * 64 * 128 * 128;        // 16.8M elems: max activation
  float* P0 = ws;
  float* P1 = ws + PSL;
  float* stats = ws + 2 * PSL;                  // 128 floats for GN stats
  float* scr0 = ws + 2 * PSL + 256;
  float* sp = scr0;
  auto alloc = [&](long n) { float* r = sp; sp += (n + 63) & ~63L; return r; };

  auto conv = [&](const float* in_, const float* w_, float* out_,
                  int Ci, int H, int W, int Co, int KW, int st, int pd) {
    int Ho = (H + 2 * pd - KW) / st + 1;
    int Wo = (W + 2 * pd - KW) / st + 1;
    dim3 grd((unsigned)((Ho * Wo + TN2 - 1) / TN2), (unsigned)((Co + TM2 - 1) / TM2), NB);
    vqgan_conv_wmma<<<grd, 256, 0, stream>>>(in_, w_, out_, Ci, H, W, Co, Ho, Wo, KW, st, pd);
  };
  auto gnorm = [&](const float* x, const float* g, const float* b, const float* add,
                   float* y, int C, int HW, int mode) {
    vqgan_gn_stats<<<NB * 4, 256, 0, stream>>>(x, stats, C, HW, 4);
    long total = (long)NB * C * HW;
    vqgan_gn_apply<<<(unsigned)((total + 255) / 256), 256, 0, stream>>>(
        x, stats, g, b, add, y, C, HW, 4, mode, total);
  };
  auto resblk = [&](const float* x, float* out, int Ci, int Co, int H, int W,
                    const float* c1, const float* g1, const float* b1,
                    const float* c2, const float* g2, const float* b2,
                    const float* skip) {
    long sz = (long)NB * Co * H * W;
    float* t1 = alloc(sz);
    float* t2 = alloc(sz);
    const float* idt = x;
    if (skip) { float* t3 = alloc(sz); conv(x, skip, t3, Ci, H, W, Co, 1, 1, 0); idt = t3; }
    conv(x, c1, t1, Ci, H, W, Co, 3, 1, 1);
    gnorm(t1, g1, b1, nullptr, t2, Co, H * W, 1);        // silu(gn(.))
    conv(t2, c2, t1, Co, H, W, Co, 3, 1, 1);
    gnorm(t1, g2, b2, idt, out, Co, H * W, 3);           // silu(gn(.) + idt)
  };
  auto attn = [&](const float* x, float* out, int C, int Hh, int Ww,
                  const float* gw, const float* bw, const float* wq,
                  const float* wk, const float* wv, const float* wp) {
    int HW = Hh * Ww;
    long asz = (long)NB * C * HW;
    float* xn = alloc(asz); float* qb = alloc(asz);
    float* kb = alloc(asz); float* vb = alloc(asz);
    float* ob = alloc(asz); float* pb = alloc(asz);
    float* sc = alloc((long)NB * 2 * HW * HW);
    gnorm(x, gw, bw, nullptr, xn, C, HW, 0);
    conv(xn, wq, qb, C, Hh, Ww, C, 1, 1, 0);
    conv(xn, wk, kb, C, Hh, Ww, C, 1, 1, 0);
    conv(xn, wv, vb, C, Hh, Ww, C, 1, 1, 0);
    int d = C / 2, batch = NB * 2;
    float scale = 1.0f / sqrtf((float)(C * 2));
    // scores[i][j] = scale * sum_c q[c][i] k[c][j]   (A transposed)
    dim3 gs((unsigned)((HW + TN2 - 1) / TN2), (unsigned)((HW + TM2 - 1) / TM2), batch);
    vqgan_gemm_wmma<<<gs, 256, 0, stream>>>(qb, kb, sc, HW, HW, d, HW, HW, HW,
        (long)d * HW, (long)d * HW, (long)HW * HW, 1, 0, scale, 0);
    vqgan_softmax<<<batch * HW, 256, 0, stream>>>(sc, HW);
    // o[c][i] = sum_j v[c][j] a[i][j]   (B transposed -> TDM 2D tile)
    dim3 go((unsigned)((HW + TN2 - 1) / TN2), (unsigned)((d + TM2 - 1) / TM2), batch);
    vqgan_gemm_wmma<<<go, 256, 0, stream>>>(vb, sc, ob, d, HW, HW, HW, HW, HW,
        (long)d * HW, (long)HW * HW, (long)d * HW, 0, 1, 1.0f, 1);
    conv(ob, wp, pb, C, Hh, Ww, C, 1, 1, 0);
    vqgan_add<<<(unsigned)((asz + 255) / 256), 256, 0, stream>>>(pb, x, out, asz);
  };

  // ============================ encoder ============================
  sp = scr0;
  { float* t = alloc(16L * 32 * 128 * 128);
    conv(I(0), I(1), t, 3, 128, 128, 32, 3, 1, 1);
    gnorm(t, I(2), I(3), nullptr, P0, 32, 128 * 128, 1); }

  sp = scr0;
  { float* c = alloc(16L * 32 * 64 * 64);
    conv(P0, I(4), c, 32, 128, 128, 32, 3, 2, 1);
    resblk(c, P1, 32, 64, 64, 64, I(5), I(6), I(7), I(8), I(9), I(10), I(11)); }

  sp = scr0;
  { float* c = alloc(16L * 64 * 32 * 32);
    conv(P1, I(12), c, 64, 64, 64, 64, 3, 2, 1);
    resblk(c, P0, 64, 128, 32, 32, I(13), I(14), I(15), I(16), I(17), I(18), I(19)); }

  sp = scr0;
  resblk(P0, P1, 128, 128, 32, 32, I(20), I(21), I(22), I(23), I(24), I(25), nullptr);

  sp = scr0;
  { float* c = alloc(16L * 128 * 16 * 16);
    conv(P1, I(26), c, 128, 32, 32, 128, 3, 2, 1);
    resblk(c, P0, 128, 256, 16, 16, I(27), I(28), I(29), I(30), I(31), I(32), I(33)); }

  sp = scr0;
  attn(P0, P1, 256, 16, 16, I(34), I(35), I(36), I(37), I(38), I(39));

  sp = scr0;
  { float* z = alloc(16L * 16 * 16 * 16);
    conv(P1, I(40), z, 256, 16, 16, 16, 3, 1, 1);
    vqgan_vq<<<(4096 + 255) / 256, 256, 0, stream>>>(z, I(82), P0, 4096, 512, 16); }

  // ============================ decoder ============================
  sp = scr0;
  { float* t = alloc(16L * 256 * 16 * 16);
    conv(P0, I(41), t, 16, 16, 16, 256, 3, 1, 1);
    gnorm(t, I(42), I(43), nullptr, P1, 256, 16 * 16, 1); }

  sp = scr0;
  { long usz = 16L * 256 * 32 * 32;
    float* u = alloc(usz);
    vqgan_up2<<<(unsigned)((usz + 255) / 256), 256, 0, stream>>>(P1, u, 16, 16, usz);
    float* c = alloc(usz);
    conv(u, I(44), c, 256, 32, 32, 256, 3, 1, 1);
    float* r = alloc(16L * 128 * 32 * 32);
    resblk(c, r, 256, 128, 32, 32, I(45), I(46), I(47), I(48), I(49), I(50), I(51));
    attn(r, P0, 128, 32, 32, I(52), I(53), I(54), I(55), I(56), I(57)); }

  sp = scr0;
  { long usz = 16L * 128 * 64 * 64;
    float* u = alloc(usz);
    vqgan_up2<<<(unsigned)((usz + 255) / 256), 256, 0, stream>>>(P0, u, 32, 32, usz);
    float* c = alloc(usz);
    conv(u, I(58), c, 128, 64, 64, 128, 3, 1, 1);
    resblk(c, P1, 128, 64, 64, 64, I(59), I(60), I(61), I(62), I(63), I(64), I(65)); }

  sp = scr0;
  resblk(P1, P0, 64, 64, 64, 64, I(66), I(67), I(68), I(69), I(70), I(71), nullptr);

  sp = scr0;
  { long usz = 16L * 64 * 128 * 128;
    float* u = alloc(usz);
    vqgan_up2<<<(unsigned)((usz + 255) / 256), 256, 0, stream>>>(P0, u, 64, 64, usz);
    float* c = alloc(usz);
    conv(u, I(72), c, 64, 128, 128, 64, 3, 1, 1);
    resblk(c, P1, 64, 32, 128, 128, I(73), I(74), I(75), I(76), I(77), I(78), I(79)); }

  sp = scr0;
  { long osz = 16L * 3 * 128 * 128;
    float* c = alloc(osz);
    conv(P1, I(80), c, 32, 128, 128, 3, 3, 1, 1);
    vqgan_bias_sigmoid<<<(unsigned)((osz + 255) / 256), 256, 0, stream>>>(
        c, I(81), (float*)d_out, 3, 128 * 128, osz); }
}
